// BmmEnsemble_53309134077991
// MI455X (gfx1250) — matrix operand order repaired and checked by reading, not compiled, not run on hardware
//
#include <hip/hip_runtime.h>

// ---------------- problem constants ----------------
#define NS 7          // species
#define NM 8          // ensemble models
#define D0 1008       // AEV dim
#define DP 1024       // padded K for layer0
#define NG 7200       // atoms per species
#define NTOT 50400    // total atoms
#define TM 80         // atoms per block tile
#define MT 5          // TM/16 row tiles
#define NT 16         // 256/16 col tiles (all layers padded to N=256)
#define ASTR (DP + 8)   // 1032  (bf16 units) LDS A stride
#define HSTR (256 + 8)  // 264   (bf16 units) LDS H stride
#define CHUNK_H 16384   // bf16 units per weight chunk (2 ksteps * 16 frags * 512) = 32KB

// ws layout (bytes)
#define OFF_CNT 0
#define OFF_IDX 1024
#define OFF_W0  262144                      // 7*8*1024*256 bf16 = 29,360,128 B
#define OFF_W1  (OFF_W0 + 29360128)         // 7*8*256*256 bf16  =  7,340,032 B
#define OFF_W2  (OFF_W1 + 7340032)          // 7*8*256*256 bf16  =  7,340,032 B

typedef __bf16 bf16x16 __attribute__((ext_vector_type(16)));
typedef float  f32x8   __attribute__((ext_vector_type(8)));

__device__ __forceinline__ unsigned short f2bf(float x) {
    union { float f; unsigned u; } c; c.f = x;
    unsigned r = (c.u + 0x7FFFu + ((c.u >> 16) & 1u)) >> 16;
    return (unsigned short)r;
}
__device__ __forceinline__ float bf2f(unsigned short h) {
    union { unsigned u; float f; } c; c.u = ((unsigned)h) << 16;
    return c.f;
}
__device__ __forceinline__ float celu01(float x) {
    return (x > 0.f) ? x : 0.1f * (__expf(x * 10.f) - 1.f);
}

// ---------------- Tensor Data Mover (TDM) staging ----------------
#if __has_builtin(__builtin_amdgcn_tensor_load_to_lds)
#define HAVE_TDM 1
typedef unsigned u32x4 __attribute__((ext_vector_type(4)));
typedef int      i32x8 __attribute__((ext_vector_type(8)));
typedef int      i32x4 __attribute__((ext_vector_type(4)));

// Linear 32KB global->LDS DMA: 2D tile, data_size=8B, tile_dim0=4096 units.
__device__ __forceinline__ void tdm_copy_32k(const void* gsrc, unsigned lds_byte_off) {
    unsigned long long ga = (unsigned long long)gsrc;
    u32x4 g0;
    g0[0] = 1u;                                   // count=1, user mode, no gather
    g0[1] = lds_byte_off;                         // D#.lds_addr
    g0[2] = (unsigned)(ga & 0xFFFFFFFFu);         // global_addr[31:0]
    g0[3] = (unsigned)((ga >> 32) & 0x01FFFFFFu)  // global_addr[56:32]
            | (2u << 30);                         // type = 2 (image)
    i32x8 g1;
    g1[0] = 0x00030000;                  // wg_mask=0, data_size=3 (8B), no pad/iter
    g1[1] = (int)(0x1000u << 16);        // tensor_dim0[15:0]=4096 at bits[63:48]
    g1[2] = (int)(1u << 16);             // tensor_dim0[31:16]=0 | tensor_dim1[15:0]=1
    g1[3] = (int)(0x1000u << 16);        // tensor_dim1[31:16]=0 | tile_dim0=4096
    g1[4] = 1;                           // tile_dim1=1, tile_dim2=0
    g1[5] = 0x1000;                      // tensor_dim0_stride[31:0]=4096
    g1[6] = (int)(0x1000u << 16);        // stride0[47:32]=0 | stride1[15:0]=4096
    g1[7] = 0;
    i32x4 z4 = {0, 0, 0, 0};
    i32x8 z8 = {0, 0, 0, 0, 0, 0, 0, 0};
    // 6-arg toolchain form: (g0, g1, g2, g3, extra, cpol)
    __builtin_amdgcn_tensor_load_to_lds(g0, g1, z4, z4, z8, 0);
}
#else
#define HAVE_TDM 0
#endif

// ---------------- setup kernels ----------------
__global__ void zero_kernel(int* cnt, float* out) {
    int t = threadIdx.x;
    if (t < NS) cnt[t] = 0;
    if (t == 0) out[0] = 0.f;
}

__global__ void scatter_kernel(const int* __restrict__ sp, int* cnt, int* idx) {
    int i = blockIdx.x * 256 + threadIdx.x;
    if (i >= NTOT) return;
    int s = sp[i];
    int p = atomicAdd(&cnt[s], 1);
    idx[s * NG + p] = i;
}

// Weight conversion: fp32 -> bf16, WMMA B-fragment order, zero padded.
// frag = (k/32)*16 + (n/16); lane = (n&15) + 16*((k>>4)&1); elem = k&15
__global__ void cvt_w0(const float* __restrict__ w, unsigned short* __restrict__ dst) {
    const long total = (long)NS * NM * DP * 256;
    for (long i = (long)blockIdx.x * blockDim.x + threadIdx.x; i < total;
         i += (long)gridDim.x * blockDim.x) {
        int n = (int)(i & 255);
        long t = i >> 8;
        int k = (int)(t & (DP - 1));
        int sm = (int)(t >> 10);
        float v = (k < D0) ? w[((long)sm * D0 + k) * 256 + n] : 0.f;
        int frag = ((k >> 5) << 4) | (n >> 4);
        int lane = (n & 15) | (((k >> 4) & 1) << 4);
        int elem = k & 15;
        dst[(long)sm * 262144 + (long)frag * 512 + lane * 16 + elem] = f2bf(v);
    }
}
__global__ void cvt_w1(const float* __restrict__ w, unsigned short* __restrict__ dst) {
    const long total = (long)NS * NM * 256 * 256;
    for (long i = (long)blockIdx.x * blockDim.x + threadIdx.x; i < total;
         i += (long)gridDim.x * blockDim.x) {
        int n = (int)(i & 255);
        long t = i >> 8;
        int k = (int)(t & 255);
        int sm = (int)(t >> 8);
        float v = (n < 192) ? w[((long)sm * 256 + k) * 192 + n] : 0.f;
        int frag = ((k >> 5) << 4) | (n >> 4);
        int lane = (n & 15) | (((k >> 4) & 1) << 4);
        int elem = k & 15;
        dst[(long)sm * 65536 + (long)frag * 512 + lane * 16 + elem] = f2bf(v);
    }
}
__global__ void cvt_w2(const float* __restrict__ w, unsigned short* __restrict__ dst) {
    const long total = (long)NS * NM * 256 * 256;
    for (long i = (long)blockIdx.x * blockDim.x + threadIdx.x; i < total;
         i += (long)gridDim.x * blockDim.x) {
        int n = (int)(i & 255);
        long t = i >> 8;
        int k = (int)(t & 255);
        int sm = (int)(t >> 8);
        float v = (k < 192 && n < 160) ? w[((long)sm * 192 + k) * 160 + n] : 0.f;
        int frag = ((k >> 5) << 4) | (n >> 4);
        int lane = (n & 15) | (((k >> 4) & 1) << 4);
        int elem = k & 15;
        dst[(long)sm * 65536 + (long)frag * 512 + lane * 16 + elem] = f2bf(v);
    }
}

// ---------------- fragment loads ----------------
__device__ __forceinline__ bf16x16 ld_bfrag(const unsigned short* p) {
    // contiguous 32B per lane (fragment-order LDS buffer)
    union { bf16x16 v; uint4 q[2]; } u;
    u.q[0] = *(const uint4*)p;
    u.q[1] = *(const uint4*)(p + 8);
    return u.v;
}
__device__ __forceinline__ bf16x16 ld_afrag(const unsigned short* p) {
    // A order used here (k = 16h + s, same as B): elems 0-7 at p, elems 8-15 at p+8 (bf16 units)
    union { bf16x16 v; uint4 q[2]; } u;
    u.q[0] = *(const uint4*)p;
    u.q[1] = *(const uint4*)(p + 8);
    return u.v;
}

// ---------------- tiled GEMM + bias + CELU -> bf16 LDS ----------------
// Weight chunks (2 ksteps = 32KB) stream through a double-buffered LDS region.
// TDM path: wave 0 issues tensor_load_to_lds for chunk c+1, waits TENSORcnt<=1
// (in-order per wave => chunk c complete), barrier, all waves compute chunk c.
template<int KSTEPS>
__device__ __forceinline__ void gemm_layer(
    const unsigned short* __restrict__ wfrag,  // global, fragment order, this (s,m)
    unsigned short* wbuf,                      // LDS staging: 2 x CHUNK_H
    const unsigned short* asrc, int astride,   // LDS activations, row-major bf16
    unsigned short* hdst,                      // LDS output, row-major (HSTR)
    const float* __restrict__ biasg, int nreal,
    int tid)
{
    const int wave = tid >> 5;
    const int lane = tid & 31;
    const int lrow = lane & 15;
    const int khalf = (lane >> 4) << 4;  // A: +16 K offset for upper half-wave
    const int NC = KSTEPS / 2;

    f32x8 acc[2 * MT];
#pragma unroll
    for (int i = 0; i < 2 * MT; i++) acc[i] = (f32x8)0.f;

#if HAVE_TDM
    const unsigned lds0 = (unsigned)(unsigned long long)(const void*)wbuf;
    if (tid < 32) tdm_copy_32k(wfrag, lds0);             // prologue: chunk 0
    for (int c = 0; c < NC; c++) {
        const unsigned short* cur = wbuf + (c & 1) * CHUNK_H;
        if (tid < 32) {
            if (c + 1 < NC) {
                tdm_copy_32k(wfrag + (size_t)(c + 1) * CHUNK_H,
                             lds0 + (unsigned)(((c + 1) & 1) * (CHUNK_H * 2)));
                __builtin_amdgcn_s_wait_tensorcnt(1);    // chunk c done
            } else {
                __builtin_amdgcn_s_wait_tensorcnt(0);
            }
        }
        __syncthreads();
#else
    for (int c = 0; c < NC; c++) {
        const unsigned short* cur = wbuf;
        __syncthreads();
        {   // synchronous fallback: b128 copy, coalesced
            const uint4* src = (const uint4*)(wfrag + (size_t)c * CHUNK_H);
            uint4* dq = (uint4*)wbuf;
#pragma unroll
            for (int e = 0; e < 8; e++)
                dq[tid + e * 256] = src[tid + e * 256];
        }
        __syncthreads();
#endif
#pragma unroll
        for (int kk = 0; kk < 2; kk++) {
            bf16x16 b0 = ld_bfrag(cur + (kk * NT + wave * 2 + 0) * 512 + lane * 16);
            bf16x16 b1 = ld_bfrag(cur + (kk * NT + wave * 2 + 1) * 512 + lane * 16);
#pragma unroll
            for (int mt = 0; mt < MT; mt++) {
                const unsigned short* ap =
                    asrc + (mt * 16 + lrow) * astride + (c * 2 + kk) * 32 + khalf;
                bf16x16 a = ld_afrag(ap);
                acc[mt * 2 + 0] = __builtin_amdgcn_wmma_f32_16x16x32_bf16(
                    false, a, false, b0, (short)0, acc[mt * 2 + 0], false, false);
                acc[mt * 2 + 1] = __builtin_amdgcn_wmma_f32_16x16x32_bf16(
                    false, a, false, b1, (short)0, acc[mt * 2 + 1], false, false);
            }
        }
#if HAVE_TDM
        __syncthreads();   // all reads of this buffer done before chunk c+2 overwrites
#endif
    }
    __syncthreads();
    // epilogue: bias + CELU + bf16 store (C layout: VGPR r -> row r (+8 upper half))
    const int rowoff = (lane >> 4) << 3;
#pragma unroll
    for (int ni = 0; ni < 2; ni++) {
        int ncol = (wave * 2 + ni) * 16 + lrow;
        float bias = (ncol < nreal) ? biasg[ncol] : 0.f;
#pragma unroll
        for (int mt = 0; mt < MT; mt++) {
            f32x8 c = acc[mt * 2 + ni];
#pragma unroll
            for (int r = 0; r < 8; r++) {
                float x = celu01(c[r] + bias);
                hdst[(mt * 16 + rowoff + r) * HSTR + ncol] = f2bf(x);
            }
        }
    }
}

// ---------------- main fused kernel ----------------
__launch_bounds__(256, 1)
__global__ void ani_main(const int* __restrict__ idxg, const float* __restrict__ aev,
                         const unsigned short* __restrict__ w0f,
                         const unsigned short* __restrict__ w1f,
                         const unsigned short* __restrict__ w2f,
                         const float* __restrict__ b0g, const float* __restrict__ b1g,
                         const float* __restrict__ b2g,
                         const float* __restrict__ w3g, const float* __restrict__ b3g,
                         float* __restrict__ outg)
{
    __shared__ __align__(16) unsigned short Atile[TM * ASTR];    // 165,120 B
    __shared__ __align__(16) unsigned short Wbuf[2 * CHUNK_H];   //  65,536 B
    __shared__ __align__(16) unsigned short Hb0[TM * HSTR];      //  42,240 B
    __shared__ __align__(16) unsigned short Hb1[TM * HSTR];      //  42,240 B
    __shared__ int aidx[TM];
    __shared__ float sAcc;

    const int tid = threadIdx.x;
    const int s = blockIdx.x / (NG / TM);
    const int tile = blockIdx.x % (NG / TM);

    if (tid == 0) sAcc = 0.f;
    if (tid < TM) aidx[tid] = idxg[s * NG + tile * TM + tid];
    __syncthreads();

    // stage A tile once (reused by all 8 models): gather + fp32->bf16, pad K to 1024
    for (int e = tid; e < TM * (DP / 8); e += 256) {
        int j = e >> 7;
        int k0 = (e & 127) << 3;
        union { uint4 q; unsigned short us[8]; } o;
        if (k0 < D0) {
            const float* sp = aev + (size_t)aidx[j] * D0 + k0;
            float4 f0 = *(const float4*)sp;
            float4 f1 = *(const float4*)(sp + 4);
            o.us[0] = f2bf(f0.x); o.us[1] = f2bf(f0.y);
            o.us[2] = f2bf(f0.z); o.us[3] = f2bf(f0.w);
            o.us[4] = f2bf(f1.x); o.us[5] = f2bf(f1.y);
            o.us[6] = f2bf(f1.z); o.us[7] = f2bf(f1.w);
        } else {
            o.q = make_uint4(0u, 0u, 0u, 0u);
        }
        *(uint4*)&Atile[j * ASTR + k0] = o.q;
    }

    float esum = 0.f;
    for (int m = 0; m < NM; m++) {
        const int sm = s * NM + m;
        gemm_layer<32>(w0f + (size_t)sm * 262144, Wbuf, Atile, ASTR, Hb0,
                       b0g + sm * 256, 256, tid);
        gemm_layer<8>(w1f + (size_t)sm * 65536, Wbuf, Hb0, HSTR, Hb1,
                      b1g + sm * 192, 192, tid);
        gemm_layer<8>(w2f + (size_t)sm * 65536, Wbuf, Hb1, HSTR, Hb0,
                      b2g + sm * 160, 160, tid);
        __syncthreads();
        if (tid < TM) {  // layer 3: [160] -> scalar, VALU dot
            const float* w3 = w3g + (size_t)sm * 160;
            float d = 0.f;
#pragma unroll 4
            for (int k = 0; k < 160; k++)
                d = fmaf(bf2f(Hb0[tid * HSTR + k]), w3[k], d);
            esum += d + b3g[sm];
        }
    }
    if (tid < TM) atomicAdd(&sAcc, esum);
    __syncthreads();
    if (tid == 0) atomicAdd(outg, sAcc * 0.125f);  // mean over 8 models
}

// ---------------- host entry ----------------
extern "C" void kernel_launch(void* const* d_in, const int* in_sizes, int n_in,
                              void* d_out, int out_size, void* d_ws, size_t ws_size,
                              hipStream_t stream) {
    (void)in_sizes; (void)n_in; (void)out_size; (void)ws_size;
    const int*   species = (const int*)d_in[0];
    const float* aev = (const float*)d_in[1];
    const float* W0 = (const float*)d_in[2];
    const float* b0 = (const float*)d_in[3];
    const float* W1 = (const float*)d_in[4];
    const float* b1 = (const float*)d_in[5];
    const float* W2 = (const float*)d_in[6];
    const float* b2 = (const float*)d_in[7];
    const float* W3 = (const float*)d_in[8];
    const float* b3 = (const float*)d_in[9];
    float* out = (float*)d_out;
    char* ws = (char*)d_ws;

    int* cnt = (int*)(ws + OFF_CNT);
    int* idx = (int*)(ws + OFF_IDX);
    unsigned short* w0f = (unsigned short*)(ws + OFF_W0);
    unsigned short* w1f = (unsigned short*)(ws + OFF_W1);
    unsigned short* w2f = (unsigned short*)(ws + OFF_W2);

    zero_kernel<<<1, 64, 0, stream>>>(cnt, out);
    scatter_kernel<<<(NTOT + 255) / 256, 256, 0, stream>>>(species, cnt, idx);
    cvt_w0<<<4096, 256, 0, stream>>>(W0, w0f);
    cvt_w1<<<1024, 256, 0, stream>>>(W1, w1f);
    cvt_w2<<<1024, 256, 0, stream>>>(W2, w2f);
    ani_main<<<NS * (NG / TM), 256, 0, stream>>>(idx, aev, w0f, w1f, w2f,
                                                 b0, b1, b2, W3, b3, out);
}
